// LinearCatVAE_19078244729235
// MI455X (gfx1250) — matrix operand order repaired and checked by reading, not compiled, not run on hardware
//
#include <hip/hip_runtime.h>
#include <cstddef>
#include <cstdint>

// ---------------------------------------------------------------------------
// LinearCatVAE loss for MI455X (gfx1250, wave32, WMMA).
//
// Roofline reasoning:
//  * After folding Psi into the encoder (E = enc_W @ Psi), the dominant op is
//    logits = eta @ Psi (4096^3, ~137 GFLOP) -> compute-bound at bf16 WMMA.
//  * GEMM uses LDS-staged 128x128 block tiles (double-buffered, 40KB of the
//    320KB/WGP LDS) shared by 8 waves; each wave register-blocks 4x2 16x16
//    accumulators -> 8 v_wmma_f32_16x16x32_bf16 per K-step, ~64 FLOP/byte of
//    global traffic per block.
//  * Epilogue is templated on MODE so each instantiation carries only its own
//    store path (previous runtime-mode version spilled the 64 accumulator
//    VGPRs to scratch at loop exit).
//  * Numerically delicate parts (Cholesky 256x256, triangular solve,
//    lgamma / log-softmax reductions, final combine) stay fp32 VALU.
// ---------------------------------------------------------------------------

typedef __bf16 bf16_t;
typedef __attribute__((ext_vector_type(16))) __bf16 v16bf;
typedef __attribute__((ext_vector_type(8)))  float  v8f;

#define BB   4096   // batch
#define DD   4096   // categories (also padded n)
#define NN   4095   // n = D-1
#define HIDD 256    // hidden
#define LDSW 40     // LDS panel row pitch in bf16 elements (32 data + 8 pad)
#define LOG2PI_F 1.8378770664093453f

// --------------------------- small helpers ---------------------------------

__global__ void zero_scalars_k(float* p) {
  if (threadIdx.x < 16) p[threadIdx.x] = 0.f;
}

// per-row stats of x: mean(log(x+1)), sum(x), sum(lgamma(x+1))
__global__ __launch_bounds__(256)
void row_x_stats(const float* __restrict__ x, float* rowmean, float* ntot, float* sumlg) {
  const int row = blockIdx.x, t = threadIdx.x;
  const float* xr = x + (size_t)row * DD;
  float slog = 0.f, sn = 0.f, slg = 0.f;
  for (int j = t; j < DD; j += 256) {
    float xv = xr[j];
    slog += logf(xv + 1.f);
    sn   += xv;
    slg  += lgammaf(xv + 1.f);
  }
  __shared__ float s1[256], s2[256], s3[256];
  s1[t] = slog; s2[t] = sn; s3[t] = slg;
  __syncthreads();
  for (int off = 128; off; off >>= 1) {
    if (t < off) { s1[t] += s1[t+off]; s2[t] += s2[t+off]; s3[t] += s3[t+off]; }
    __syncthreads();
  }
  if (t == 0) { rowmean[row] = s1[0] / (float)DD; ntot[row] = s2[0]; sumlg[row] = s3[0]; }
}

// clr = log(x+1) - rowmean  ->  bf16
__global__ __launch_bounds__(256)
void clr_kernel(const float* __restrict__ x, const float* __restrict__ rowmean,
                bf16_t* __restrict__ clr) {
  size_t i = (size_t)blockIdx.x * blockDim.x + threadIdx.x;
  if (i >= (size_t)BB * DD) return;
  int row = (int)(i >> 12);                  // DD == 4096
  clr[i] = (bf16_t)(logf(x[i] + 1.f) - rowmean[row]);
}

// dst[r][c] = (r<srcR && c<srcC) ? (bf16)src[r][c] : 0   (zero-pad copy)
__global__ __launch_bounds__(256)
void copy_pad_bf16(const float* __restrict__ src, int srcR, int srcC,
                   bf16_t* __restrict__ dst, int dstR, int dstC) {
  size_t i = (size_t)blockIdx.x * blockDim.x + threadIdx.x;
  size_t total = (size_t)dstR * dstC;
  if (i >= total) return;
  int r = (int)(i / dstC), c = (int)(i % dstC);
  float v = (r < srcR && c < srcC) ? src[(size_t)r * srcC + c] : 0.f;
  dst[i] = (bf16_t)v;
}

// dst[r][c] = (c<srcR && r<srcC) ? (bf16)src[c][r] : 0   (LDS-tiled transpose)
__global__ void transpose_pad_bf16(const float* __restrict__ src, int srcR, int srcC,
                                   bf16_t* __restrict__ dst, int dstR, int dstC) {
  __shared__ float tile[16][17];
  int tx = threadIdx.x, ty = threadIdx.y;
  int sr = blockIdx.x * 16 + ty;   // src row  (== dst col block)
  int sc = blockIdx.y * 16 + tx;   // src col  (== dst row block)
  tile[ty][tx] = (sr < srcR && sc < srcC) ? src[(size_t)sr * srcC + sc] : 0.f;
  __syncthreads();
  int r = blockIdx.y * 16 + ty;    // dst row
  int c = blockIdx.x * 16 + tx;    // dst col
  if (r < dstR && c < dstC) dst[(size_t)r * dstC + c] = (bf16_t)tile[tx][ty];
}

// --------------------------- WMMA GEMM -------------------------------------
// C(MxN) = A(MxK) * B(KxN); A row-major bf16, BT = B^T (NxK) row-major bf16.
// Requirements: M % 128 == 0, N % 128 == 0, K % 32 == 0 (all call sites obey).
// Workgroup: 256 threads (8 waves); 128x128 block tile; each wave computes a
// 64x32 region as 4x2 accumulators; LDS double-buffered A/BT panels.
// MODE bit0: write f32 C; bit1: write bf16 Cb; bit2: v = aux[row][col] - acc
//           (aux fp32 with ldaux cols, out-of-range aux col -> 0).
template <int MODE>
__global__ __launch_bounds__(256)
void gemm_bf16_wmma(const bf16_t* __restrict__ A, int lda,
                    const bf16_t* __restrict__ BT, int ldbt,
                    float* __restrict__ C, bf16_t* __restrict__ Cb, int ldc,
                    const float* __restrict__ aux, int ldaux, int K) {
  __shared__ alignas(16) bf16_t lsA[2][128 * LDSW];
  __shared__ alignas(16) bf16_t lsB[2][128 * LDSW];

  const int tid  = threadIdx.x;
  const int lane = tid & 31;
  const int wave = tid >> 5;
  const int wm   = wave >> 2;          // 0..1 : which 64-row slab
  const int wn   = wave & 3;           // 0..3 : which 32-col slab

  const bf16_t* gA  = A  + (size_t)blockIdx.y * 128 * lda;
  const bf16_t* gBT = BT + (size_t)blockIdx.x * 128 * ldbt;

  // staging map: 512 16B chunks per panel; thread t handles chunks t and t+256
  const int c0 = tid, c1 = tid + 256;
  const int ar0 = c0 >> 2, ac0 = (c0 & 3) * 8;
  const int ar1 = c1 >> 2, ac1 = (c1 & 3) * 8;
  const size_t gaOff0 = (size_t)ar0 * lda  + ac0;
  const size_t gaOff1 = (size_t)ar1 * lda  + ac1;
  const size_t gbOff0 = (size_t)ar0 * ldbt + ac0;
  const size_t gbOff1 = (size_t)ar1 * ldbt + ac1;
  const int lsOff0 = ar0 * LDSW + ac0;
  const int lsOff1 = ar1 * LDSW + ac1;

  // fragment read geometry (CDNA5 WMMA bf16 layouts)
  const int aRow = (lane & 15), aK = (lane >> 4) * 8;   // A: lane-half splits K {0..7,16..23}/{8..15,24..31}
  const int bCol = (lane & 15), bK = (lane >> 4) * 8;   // B: lane-half splits K {0..7,16..23}/{8..15,24..31}

  v8f zero = {0.f, 0.f, 0.f, 0.f, 0.f, 0.f, 0.f, 0.f};
  v8f acc[4][2];
  for (int i = 0; i < 4; ++i)
    for (int j = 0; j < 2; ++j)
      acc[i][j] = zero;

  // stage k = 0 into buffer 0
  {
    uint4 ta0 = *(const uint4*)(gA  + gaOff0);
    uint4 ta1 = *(const uint4*)(gA  + gaOff1);
    uint4 tb0 = *(const uint4*)(gBT + gbOff0);
    uint4 tb1 = *(const uint4*)(gBT + gbOff1);
    *(uint4*)&lsA[0][lsOff0] = ta0;
    *(uint4*)&lsA[0][lsOff1] = ta1;
    *(uint4*)&lsB[0][lsOff0] = tb0;
    *(uint4*)&lsB[0][lsOff1] = tb1;
  }
  __syncthreads();

  int cur = 0;
  for (int k = 0; k < K; k += 32) {
    const bool hasNext = (k + 32) < K;
    uint4 ta0 = {0,0,0,0}, ta1 = {0,0,0,0}, tb0 = {0,0,0,0}, tb1 = {0,0,0,0};
    if (hasNext) {
      const int kn = k + 32;
      ta0 = *(const uint4*)(gA  + kn + gaOff0);
      ta1 = *(const uint4*)(gA  + kn + gaOff1);
      tb0 = *(const uint4*)(gBT + kn + gbOff0);
      tb1 = *(const uint4*)(gBT + kn + gbOff1);
      __builtin_prefetch(gA  + kn + 32 + gaOff0, 0, 1);   // global_prefetch_b8
      __builtin_prefetch(gBT + kn + 32 + gbOff0, 0, 1);
    }

    // fragments from LDS (ds_load_b128), shared by all 8 waves
    const bf16_t* pa = lsA[cur];
    const bf16_t* pb = lsB[cur];
    union { v16bf v; uint4 q[2]; } af[4], bfr[2];
    for (int mt = 0; mt < 4; ++mt) {
      int off = (wm * 64 + mt * 16 + aRow) * LDSW + aK;
      af[mt].q[0] = *(const uint4*)(pa + off);
      af[mt].q[1] = *(const uint4*)(pa + off + 16);
    }
    for (int nt = 0; nt < 2; ++nt) {
      int off = (wn * 32 + nt * 16 + bCol) * LDSW + bK;
      bfr[nt].q[0] = *(const uint4*)(pb + off);
      bfr[nt].q[1] = *(const uint4*)(pb + off + 16);
    }
    for (int mt = 0; mt < 4; ++mt)
      for (int nt = 0; nt < 2; ++nt)
        acc[mt][nt] = __builtin_amdgcn_wmma_f32_16x16x32_bf16(
            false, af[mt].v, false, bfr[nt].v, (short)0, acc[mt][nt], false, false);

    if (hasNext) {
      int nxt = cur ^ 1;
      *(uint4*)&lsA[nxt][lsOff0] = ta0;
      *(uint4*)&lsA[nxt][lsOff1] = ta1;
      *(uint4*)&lsB[nxt][lsOff0] = tb0;
      *(uint4*)&lsB[nxt][lsOff1] = tb1;
    }
    __syncthreads();
    cur ^= 1;
  }

  // epilogue: C element r of each tile -> row 8*(lane>>4)+r, col lane&15
  const int baseRow = blockIdx.y * 128 + wm * 64 + (lane >> 4) * 8;
  const int baseCol = blockIdx.x * 128 + wn * 32 + (lane & 15);
  for (int mt = 0; mt < 4; ++mt) {
    for (int nt = 0; nt < 2; ++nt) {
      const int col = baseCol + nt * 16;
      for (int r = 0; r < 8; ++r) {
        const int row = baseRow + mt * 16 + r;
        float v = acc[mt][nt][r];
        if constexpr (MODE & 4) {
          v = (col < ldaux) ? (aux[(size_t)row * ldaux + col] - v) : 0.f;
        }
        size_t idx = (size_t)row * ldc + col;
        if constexpr (MODE & 1) C[idx]  = v;
        if constexpr (MODE & 2) Cb[idx] = (bf16_t)v;
      }
    }
  }
}

// --------------------------- solve path ------------------------------------

// Mm = diag(exp(-lv)) + WtW / var
__global__ void build_M(const float* __restrict__ WtW, const float* __restrict__ lv,
                        const float* __restrict__ lss, float* __restrict__ Mm) {
  int i = blockIdx.x, j = threadIdx.x;
  float var = expf(lss[0]);
  float v = WtW[i * HIDD + j] / var;
  if (i == j) v += expf(-lv[i]);
  Mm[i * HIDD + j] = v;
}

// In-place lower Cholesky of 256x256 SPD matrix; logdetOut = log det M.
__global__ __launch_bounds__(256)
void cholesky256(float* __restrict__ Mm, float* __restrict__ logdetOut) {
  __shared__ float colk[HIDD];
  __shared__ float diag;
  const int t = threadIdx.x;
  float ld = 0.f;
  for (int k = 0; k < HIDD; ++k) {
    if (t == 0) { diag = sqrtf(Mm[k * HIDD + k]); Mm[k * HIDD + k] = diag; }
    __syncthreads();
    float Lkk = diag;
    if (t > k) { float v = Mm[t * HIDD + k] / Lkk; Mm[t * HIDD + k] = v; colk[t] = v; }
    if (t == 0) ld += 2.f * logf(Lkk);
    __syncthreads();
    if (t > k) {
      float lik = colk[t];
      for (int j = k + 1; j <= t; ++j) Mm[t * HIDD + j] -= lik * colk[j];
    }
    __syncthreads();
  }
  if (t == 0) *logdetOut = ld;
}

// q2[b] = dW_b^T M^{-1} dW_b = || L^{-1} dW_b ||^2  (forward substitution)
__global__ __launch_bounds__(256)
void fsub_quad(const float* __restrict__ dW, const float* __restrict__ Lm,
               float* __restrict__ q2) {
  const int b = blockIdx.x, t = threadIdx.x;
  __shared__ float y[HIDD];
  __shared__ float yk;
  y[t] = dW[(size_t)b * HIDD + t];
  __syncthreads();
  for (int k = 0; k < HIDD; ++k) {
    if (t == k) { float v = y[k] / Lm[k * HIDD + k]; y[k] = v; yk = v; }
    __syncthreads();
    if (t > k) y[t] -= Lm[t * HIDD + k] * yk;
    __syncthreads();
  }
  float v = y[t]; v *= v;
  __shared__ float s[HIDD];
  s[t] = v;
  __syncthreads();
  for (int off = 128; off; off >>= 1) {
    if (t < off) s[t] += s[t + off];
    __syncthreads();
  }
  if (t == 0) q2[b] = s[0];
}

// ddot[b] = sum_j d[b][j]^2  (d stored bf16, fp32 accumulate)
__global__ __launch_bounds__(256)
void ddot_kernel(const bf16_t* __restrict__ dB, float* __restrict__ ddot) {
  const int row = blockIdx.x, t = threadIdx.x;
  const bf16_t* dr = dB + (size_t)row * DD;
  float sum = 0.f;
  for (int j = t; j < DD; j += 256) { float v = (float)dr[j]; sum += v * v; }
  __shared__ float s[256];
  s[t] = sum;
  __syncthreads();
  for (int off = 128; off; off >>= 1) {
    if (t < off) s[t] += s[t + off];
    __syncthreads();
  }
  if (t == 0) ddot[row] = s[0];
}

// Multinomial row term: lgamma(ntot+1) - sum lgamma(x+1) + x.logits - ntot*lse
__global__ __launch_bounds__(256)
void mult_kernel(const float* __restrict__ logits, const float* __restrict__ x,
                 const float* __restrict__ ntot, const float* __restrict__ sumlg,
                 float* __restrict__ multA) {
  const int row = blockIdx.x, t = threadIdx.x;
  const float* lr = logits + (size_t)row * DD;
  const float* xr = x + (size_t)row * DD;
  float m = -3.0e38f;
  for (int j = t; j < DD; j += 256) m = fmaxf(m, lr[j]);
  __shared__ float sm[256];
  sm[t] = m;
  __syncthreads();
  for (int off = 128; off; off >>= 1) {
    if (t < off) sm[t] = fmaxf(sm[t], sm[t + off]);
    __syncthreads();
  }
  float rowmax = sm[0];
  __syncthreads();
  float se = 0.f, xd = 0.f;
  for (int j = t; j < DD; j += 256) {
    float l = lr[j];
    se += expf(l - rowmax);
    xd += xr[j] * l;
  }
  __shared__ float s1[256], s2[256];
  s1[t] = se; s2[t] = xd;
  __syncthreads();
  for (int off = 128; off; off >>= 1) {
    if (t < off) { s1[t] += s1[t+off]; s2[t] += s2[t+off]; }
    __syncthreads();
  }
  if (t == 0) {
    float lse = rowmax + logf(s1[0]);
    float nt = ntot[row];
    multA[row] = lgammaf(nt + 1.f) - sumlg[row] + s2[0] - nt * lse;
  }
}

// sum of z_mean^2 (atomic accumulate into priorSum)
__global__ __launch_bounds__(256)
void prior_kernel(const float* __restrict__ zm, float* __restrict__ priorSum) {
  size_t i = (size_t)blockIdx.x * blockDim.x + threadIdx.x;
  const size_t total = (size_t)BB * HIDD;
  float v = 0.f;
  for (size_t idx = i; idx < total; idx += (size_t)gridDim.x * blockDim.x) {
    float z = zm[idx]; v += z * z;
  }
  __shared__ float s[256];
  s[threadIdx.x] = v;
  __syncthreads();
  for (int off = 128; off; off >>= 1) {
    if (threadIdx.x < off) s[threadIdx.x] += s[threadIdx.x + off];
    __syncthreads();
  }
  if (threadIdx.x == 0) atomicAdd(priorSum, s[0]);
}

// Final scalar combine
__global__ __launch_bounds__(1024)
void final_kernel(const float* __restrict__ multA, const float* __restrict__ ddot,
                  const float* __restrict__ q2, const float* __restrict__ lv,
                  const float* __restrict__ lss_p, const float* __restrict__ logdetM,
                  const float* __restrict__ priorSum, float* __restrict__ out) {
  const int t = threadIdx.x;
  float sM = 0.f, sD = 0.f, sQ = 0.f, sL = 0.f;
  for (int b = t; b < BB; b += 1024) { sM += multA[b]; sD += ddot[b]; sQ += q2[b]; }
  if (t < HIDD) sL = lv[t];
  __shared__ float r1[1024], r2[1024], r3[1024], r4[1024];
  r1[t] = sM; r2[t] = sD; r3[t] = sQ; r4[t] = sL;
  __syncthreads();
  for (int off = 512; off; off >>= 1) {
    if (t < off) { r1[t]+=r1[t+off]; r2[t]+=r2[t+off]; r3[t]+=r3[t+off]; r4[t]+=r4[t+off]; }
    __syncthreads();
  }
  if (t == 0) {
    float lss = lss_p[0];
    float var = expf(lss);
    float nF = (float)NN;
    float quadMean = (r2[0] / var - r3[0] / (var * var)) / (float)BB;
    float logdetSigma = nF * lss + r4[0] + logdetM[0];
    float logit_loss = -0.5f * (nF * LOG2PI_F + logdetSigma) - 0.5f * quadMean;
    float prior_loss = -0.5f * priorSum[0] / ((float)BB * (float)HIDD) - 0.5f * LOG2PI_F;
    float mult_loss = r1[0] / (float)BB;
    out[0] = -(mult_loss + logit_loss + prior_loss);
  }
}

// --------------------------- launch ----------------------------------------

extern "C" void kernel_launch(void* const* d_in, const int* in_sizes, int n_in,
                              void* d_out, int out_size, void* d_ws, size_t ws_size,
                              hipStream_t stream) {
  (void)in_sizes; (void)n_in; (void)out_size; (void)ws_size;
  const float* x    = (const float*)d_in[0];   // (B, D)
  const float* Psi  = (const float*)d_in[1];   // (n, D)
  const float* encW = (const float*)d_in[2];   // (HID, n)
  const float* decW = (const float*)d_in[3];   // (n, HID)
  const float* lv   = (const float*)d_in[4];   // (HID,)
  const float* lss  = (const float*)d_in[5];   // scalar
  const float* eta  = (const float*)d_in[6];   // (B, n)
  float* out = (float*)d_out;

  char* w = (char*)d_ws;
  auto alloc = [&](size_t bytes) -> void* {
    void* p = (void*)w;
    w += (bytes + 255) & ~(size_t)255;
    return p;
  };
  bf16_t* PsiT  = (bf16_t*)alloc((size_t)DD * DD * 2);    // Psi^T, zero-padded (32MB)
  bf16_t* clrB  = (bf16_t*)alloc((size_t)BB * DD * 2);    // clr (32MB)
  bf16_t* etaB  = (bf16_t*)alloc((size_t)BB * DD * 2);    // eta padded (32MB)
  bf16_t* dB    = (bf16_t*)alloc((size_t)BB * DD * 2);    // d = eta - mu (32MB)
  float*  logit = (float*)alloc((size_t)BB * DD * 4);     // logits (64MB)
  bf16_t* encWB = (bf16_t*)alloc((size_t)HIDD * DD * 2);
  bf16_t* decWB = (bf16_t*)alloc((size_t)DD * HIDD * 2);
  bf16_t* decWT = (bf16_t*)alloc((size_t)HIDD * DD * 2);
  bf16_t* Eb    = (bf16_t*)alloc((size_t)HIDD * DD * 2);  // E = encW @ Psi
  float*  zm    = (float*)alloc((size_t)BB * HIDD * 4);
  bf16_t* zmB   = (bf16_t*)alloc((size_t)BB * HIDD * 2);
  float*  dW    = (float*)alloc((size_t)BB * HIDD * 4);
  float*  WtW   = (float*)alloc((size_t)HIDD * HIDD * 4);
  float*  Mm    = (float*)alloc((size_t)HIDD * HIDD * 4);
  float*  rowmean = (float*)alloc((size_t)BB * 4);
  float*  ntot    = (float*)alloc((size_t)BB * 4);
  float*  sumlg   = (float*)alloc((size_t)BB * 4);
  float*  ddot    = (float*)alloc((size_t)BB * 4);
  float*  q2      = (float*)alloc((size_t)BB * 4);
  float*  multA   = (float*)alloc((size_t)BB * 4);
  float*  scal    = (float*)alloc(64);   // [0]=priorSum, [1]=logdetM

  const dim3 t16(16, 16);

  zero_scalars_k<<<1, 32, 0, stream>>>(scal);
  row_x_stats<<<BB, 256, 0, stream>>>(x, rowmean, ntot, sumlg);
  clr_kernel<<<(BB * DD) / 256, 256, 0, stream>>>(x, rowmean, clrB);
  transpose_pad_bf16<<<dim3(DD/16, DD/16), t16, 0, stream>>>(Psi, NN, DD, PsiT, DD, DD);
  copy_pad_bf16<<<(HIDD * DD) / 256, 256, 0, stream>>>(encW, HIDD, NN, encWB, HIDD, DD);
  copy_pad_bf16<<<(BB * DD) / 256, 256, 0, stream>>>(eta, BB, NN, etaB, BB, DD);
  copy_pad_bf16<<<(DD * HIDD) / 256, 256, 0, stream>>>(decW, NN, HIDD, decWB, DD, HIDD);
  transpose_pad_bf16<<<dim3(DD/16, HIDD/16), t16, 0, stream>>>(decW, NN, HIDD, decWT, HIDD, DD);

  // E = encW @ Psi                       (bf16 out)
  gemm_bf16_wmma<2><<<dim3(DD/128, HIDD/128), 256, 0, stream>>>(
      encWB, DD, PsiT, DD, nullptr, Eb, DD, nullptr, 0, DD);
  // z_mean = clr @ E^T                   (f32 + bf16 out)
  gemm_bf16_wmma<3><<<dim3(HIDD/128, BB/128), 256, 0, stream>>>(
      clrB, DD, Eb, DD, zm, zmB, HIDD, nullptr, 0, DD);
  // d = eta - z_mean @ decW^T            (bf16 out, fused subtract)
  gemm_bf16_wmma<6><<<dim3(DD/128, BB/128), 256, 0, stream>>>(
      zmB, HIDD, decWB, HIDD, nullptr, dB, DD, eta, NN, HIDD);
  // dW = d @ decW                        (f32)
  gemm_bf16_wmma<1><<<dim3(HIDD/128, BB/128), 256, 0, stream>>>(
      dB, DD, decWT, DD, dW, nullptr, HIDD, nullptr, 0, DD);
  // WtW = decW^T @ decW                  (f32)
  gemm_bf16_wmma<1><<<dim3(HIDD/128, HIDD/128), 256, 0, stream>>>(
      decWT, DD, decWT, DD, WtW, nullptr, HIDD, nullptr, 0, DD);
  // logits = eta @ Psi                   (f32) -- the dominant GEMM
  gemm_bf16_wmma<1><<<dim3(DD/128, BB/128), 256, 0, stream>>>(
      etaB, DD, PsiT, DD, logit, nullptr, DD, nullptr, 0, DD);

  build_M<<<HIDD, HIDD, 0, stream>>>(WtW, lv, lss, Mm);
  cholesky256<<<1, HIDD, 0, stream>>>(Mm, scal + 1);
  fsub_quad<<<BB, HIDD, 0, stream>>>(dW, Mm, q2);
  ddot_kernel<<<BB, 256, 0, stream>>>(dB, ddot);
  mult_kernel<<<BB, 256, 0, stream>>>(logit, x, ntot, sumlg, multA);
  prior_kernel<<<256, 256, 0, stream>>>(zm, scal + 0);
  final_kernel<<<1, 1024, 0, stream>>>(multA, ddot, q2, lv, lss, scal + 1, scal + 0, out);
}
